// FloatMinGRU_70136815943789
// MI455X (gfx1250) — matrix operand order repaired and checked by reading, not compiled, not run on hardware
//
#include <hip/hip_runtime.h>
#include <math.h>

// ---------------------------------------------------------------------------
// Persistent cooperative minGRU kernel for MI455X (gfx1250, wave32, WMMA).
//   * bf16 weights staged once into d_ws (9.7MB, L2-resident on a 192MB L2)
//   * all GEMMs via v_wmma_f32_16x16x32_bf16, fp32 accumulation
//   * one kernel loops over all T=2048 steps; stages separated by an atomic
//     sense-reversal grid barrier across NWG=16 workgroups (deterministic)
// ---------------------------------------------------------------------------

#define NWG      16
#define NTHREADS 256
#define NWAVES   (NWG * (NTHREADS / 32))   // 128 waves

constexpr int Bd = 64, Td = 2048, Vd = 256, Hd = 256, Ld = 6, H4 = 1024;

typedef __attribute__((ext_vector_type(16))) __bf16 v16bf;
typedef __attribute__((ext_vector_type(8)))  float  v8f;

// ---- workspace layout (bytes) ---------------------------------------------
constexpr size_t OFF_SYNC = 0;                                   // 256B ctrl
constexpr size_t OFF_WF   = 256;
constexpr size_t SZ_HH    = (size_t)Ld * Hd * Hd * 2;            // 768KB
constexpr size_t OFF_UF   = OFF_WF + SZ_HH;
constexpr size_t OFF_WH   = OFF_UF + SZ_HH;
constexpr size_t OFF_UH   = OFF_WH + SZ_HH;
constexpr size_t SZ_MLP   = (size_t)Ld * H4 * Hd * 2;            // 3MB
constexpr size_t OFF_UP   = OFF_UH + SZ_HH;
constexpr size_t OFF_DOWN = OFF_UP + SZ_MLP;
constexpr size_t OFF_HEAD = OFF_DOWN + SZ_MLP;
constexpr size_t OFF_S    = OFF_HEAD + (size_t)Vd * Hd * 2;      // fp32 states
constexpr size_t OFF_SBF  = OFF_S + (size_t)Ld * Bd * Hd * 4;    // bf16 mirror
constexpr size_t OFF_X    = OFF_SBF + (size_t)Ld * Bd * Hd * 2;  // fp32 x
constexpr size_t OFF_XNA  = OFF_X + (size_t)Bd * Hd * 4;
constexpr size_t OFF_XNB  = OFF_XNA + (size_t)Bd * Hd * 2;
constexpr size_t OFF_XNC  = OFF_XNB + (size_t)Bd * Hd * 2;
constexpr size_t OFF_F    = OFF_XNC + (size_t)Bd * Hd * 2;       // fp32 gate
constexpr size_t OFF_HX   = OFF_F + (size_t)Bd * Hd * 4;         // fp32 xn@Wh^T
constexpr size_t OFF_Z    = OFF_HX + (size_t)Bd * Hd * 4;        // bf16 s*f
constexpr size_t OFF_U    = OFF_Z + (size_t)Bd * Hd * 2;         // bf16 mlp act

// ---- grid barrier (sense via generation counter; tolerant of any gen) ------
__device__ __forceinline__ void gbar(unsigned* sync) {
  __threadfence();
  __syncthreads();
  if (threadIdx.x == 0) {
    unsigned* cnt = sync;
    unsigned* gen = sync + 16;
    unsigned g = __hip_atomic_load(gen, __ATOMIC_ACQUIRE, __HIP_MEMORY_SCOPE_AGENT);
    __threadfence();
    unsigned prev = __hip_atomic_fetch_add(cnt, 1u, __ATOMIC_ACQ_REL, __HIP_MEMORY_SCOPE_AGENT);
    if (prev == NWG - 1) {
      __hip_atomic_store(cnt, 0u, __ATOMIC_RELAXED, __HIP_MEMORY_SCOPE_AGENT);
      __threadfence();
      __hip_atomic_fetch_add(gen, 1u, __ATOMIC_ACQ_REL, __HIP_MEMORY_SCOPE_AGENT);
    } else {
      while (__hip_atomic_load(gen, __ATOMIC_ACQUIRE, __HIP_MEMORY_SCOPE_AGENT) == g)
        __builtin_amdgcn_s_sleep(2);
    }
  }
  __syncthreads();
  __threadfence();
}

// ---- WMMA operand loaders (layouts per CDNA5 ISA sec. 7.12.2) --------------
// A: 16-bit 16x32 tile. lane m = lane&15, kh = lane>>4:
//   elems 0..7  -> K = kh*8 + j ; elems 8..15 -> K = 16 + kh*8 + j
__device__ __forceinline__ v16bf loadA(const __bf16* __restrict__ A, int lda,
                                       int m0, int k0, int lane) {
  const __bf16* p = A + (size_t)(m0 + (lane & 15)) * lda + k0 + ((lane >> 4) << 3);
  float4 lo = *(const float4*)(p);
  float4 hi = *(const float4*)(p + 16);
  v16bf a;
  __builtin_memcpy(&a, &lo, 16);
  __builtin_memcpy((char*)&a + 16, &hi, 16);
  return a;
}

// B: 32x16 tile of W^T where W is row-major [N,K]: B[k][n] = W[n][k].
// lane n = lane&15, kh = lane>>4: elems 0..7 -> K = kh*8 + e ; elems 8..15 -> K = 16 + kh*8 + (e-8)
__device__ __forceinline__ v16bf loadB(const __bf16* __restrict__ W, int ldw,
                                       int n0, int k0, int lane) {
  const __bf16* p = W + (size_t)(n0 + (lane & 15)) * ldw + k0 + ((lane >> 4) << 3);
  float4 lo = *(const float4*)(p);
  float4 hi = *(const float4*)(p + 16);
  v16bf b;
  __builtin_memcpy(&b, &lo, 16);
  __builtin_memcpy((char*)&b + 16, &hi, 16);
  return b;
}

__device__ __forceinline__ v8f mm_tile(const __bf16* __restrict__ A, int lda,
                                       const __bf16* __restrict__ W, int ldw,
                                       int m0, int n0, int K, int lane, v8f acc) {
  for (int k = 0; k < K; k += 32) {
    v16bf a = loadA(A, lda, m0, k, lane);
    v16bf b = loadB(W, ldw, n0, k, lane);
    acc = __builtin_amdgcn_wmma_f32_16x16x32_bf16(false, a, false, b,
                                                  (short)0, acc, false, false);
  }
  return acc;
}

// ---- LayerNorm: one wave handles one row of 256 (8 elems/lane) -------------
__device__ __forceinline__ void ln_row(const float* __restrict__ x,
                                       __bf16* __restrict__ out, int b, int lane) {
  float v[8]; float s = 0.f, s2 = 0.f;
#pragma unroll
  for (int j = 0; j < 8; ++j) {
    float t = x[b * Hd + lane + 32 * j];
    v[j] = t; s += t; s2 += t * t;
  }
#pragma unroll
  for (int o = 16; o > 0; o >>= 1) {
    s  += __shfl_xor(s,  o, 32);
    s2 += __shfl_xor(s2, o, 32);
  }
  float m  = s * (1.f / Hd);
  float var = s2 * (1.f / Hd) - m * m;       // biased var, matches jnp.var
  float r  = rsqrtf(var + 1e-5f);
#pragma unroll
  for (int j = 0; j < 8; ++j)
    out[b * Hd + lane + 32 * j] = (__bf16)((v[j] - m) * r);
}

__global__ void sync_init_kernel(unsigned* sync) {
  if (threadIdx.x < 64) sync[threadIdx.x] = 0u;
}

__global__ void __launch_bounds__(NTHREADS)
mingru_persistent(const int* __restrict__ tokens, const float* __restrict__ emb,
                  const float* __restrict__ W_f, const float* __restrict__ U_f,
                  const float* __restrict__ b_f, const float* __restrict__ W_h,
                  const float* __restrict__ U_h, const float* __restrict__ b_h,
                  const float* __restrict__ mlp_up, const float* __restrict__ mlp_down,
                  const float* __restrict__ head, const float* __restrict__ init_states,
                  float* __restrict__ out, char* __restrict__ ws) {
  unsigned* sync = (unsigned*)(ws + OFF_SYNC);
  __bf16* wf   = (__bf16*)(ws + OFF_WF);
  __bf16* uf   = (__bf16*)(ws + OFF_UF);
  __bf16* wh   = (__bf16*)(ws + OFF_WH);
  __bf16* uh   = (__bf16*)(ws + OFF_UH);
  __bf16* up   = (__bf16*)(ws + OFF_UP);
  __bf16* down = (__bf16*)(ws + OFF_DOWN);
  __bf16* hd   = (__bf16*)(ws + OFF_HEAD);
  float*  sbuf = (float*)(ws + OFF_S);
  __bf16* sbf  = (__bf16*)(ws + OFF_SBF);
  float*  xbuf = (float*)(ws + OFF_X);
  __bf16* xna  = (__bf16*)(ws + OFF_XNA);
  __bf16* xnb  = (__bf16*)(ws + OFF_XNB);
  __bf16* xnc  = (__bf16*)(ws + OFF_XNC);
  float*  fbuf = (float*)(ws + OFF_F);
  float*  hxb  = (float*)(ws + OFF_HX);
  __bf16* zbf  = (__bf16*)(ws + OFF_Z);
  __bf16* ubuf = (__bf16*)(ws + OFF_U);

  const int tid    = threadIdx.x;
  const int lane   = tid & 31;
  const int gw     = blockIdx.x * (NTHREADS / 32) + (tid >> 5);  // 0..127
  const int gtid   = blockIdx.x * NTHREADS + tid;
  const int stride = NWG * NTHREADS;                              // 4096

  // ---- stage -1: convert weights to bf16 (L2-resident thereafter) ---------
  for (int e = gtid; e < Ld * Hd * Hd; e += stride) {
    wf[e] = (__bf16)W_f[e]; uf[e] = (__bf16)U_f[e];
    wh[e] = (__bf16)W_h[e]; uh[e] = (__bf16)U_h[e];
  }
  for (int e = gtid; e < Ld * H4 * Hd; e += stride) {
    up[e] = (__bf16)mlp_up[e]; down[e] = (__bf16)mlp_down[e];
  }
  for (int e = gtid; e < Vd * Hd; e += stride) hd[e] = (__bf16)head[e];
  for (int e = gtid; e < Ld * Bd * Hd; e += stride) {
    float v = init_states[e]; sbuf[e] = v; sbf[e] = (__bf16)v;
  }
  gbar(sync);

  for (int t = 0; t < Td; ++t) {
    // S0: x = emb[tok]
    for (int e = gtid; e < Bd * Hd; e += stride) {
      int b = e >> 8, h = e & 255;
      xbuf[e] = emb[(size_t)tokens[b * Td + t] * Hd + h];
    }
    gbar(sync);

    for (int i = 0; i < Ld; ++i) {
      const __bf16* wfi = wf + (size_t)i * Hd * Hd;
      const __bf16* ufi = uf + (size_t)i * Hd * Hd;
      const __bf16* whi = wh + (size_t)i * Hd * Hd;
      const __bf16* uhi = uh + (size_t)i * Hd * Hd;
      const __bf16* upi = up + (size_t)i * H4 * Hd;
      const __bf16* dni = down + (size_t)i * Hd * H4;
      float* si  = sbuf + (size_t)i * Bd * Hd;
      __bf16* sbi = sbf + (size_t)i * Bd * Hd;

      // S1: xn = LN(x)
      if (gw < Bd) ln_row(xbuf, xna, gw, lane);
      gbar(sync);

      // S2: waves 0-63: f = sigmoid(xn@Wf^T + s@Uf^T + bf)
      //     waves 64-127: hx = xn@Wh^T  (bias added later)
      if (gw < 64) {
        int mt = gw >> 4, nt = gw & 15;
        v8f acc = {0.f,0.f,0.f,0.f,0.f,0.f,0.f,0.f};
        acc = mm_tile(xna, Hd, wfi, Hd, mt * 16, nt * 16, Hd, lane, acc);
        acc = mm_tile(sbi, Hd, ufi, Hd, mt * 16, nt * 16, Hd, lane, acc);
        int gn = nt * 16 + (lane & 15);
        int mb = mt * 16 + ((lane >> 4) << 3);
        float bias = b_f[i * Hd + gn];
#pragma unroll
        for (int r = 0; r < 8; ++r)
          fbuf[(mb + r) * Hd + gn] = 1.f / (1.f + expf(-(acc[r] + bias)));
      } else {
        int tt = gw - 64, mt = tt >> 4, nt = tt & 15;
        v8f acc = {0.f,0.f,0.f,0.f,0.f,0.f,0.f,0.f};
        acc = mm_tile(xna, Hd, whi, Hd, mt * 16, nt * 16, Hd, lane, acc);
        int gn = nt * 16 + (lane & 15);
        int mb = mt * 16 + ((lane >> 4) << 3);
#pragma unroll
        for (int r = 0; r < 8; ++r) hxb[(mb + r) * Hd + gn] = acc[r];
      }
      gbar(sync);

      // S3: z = s * f (bf16)
      for (int e = gtid; e < Bd * Hd; e += stride)
        zbf[e] = (__bf16)(si[e] * fbuf[e]);
      gbar(sync);

      // S4: h~ = tanh(hx + z@Uh^T + bh); ns = f*s + (1-f)*h~; s=ns; x+=ns
      if (gw < 64) {
        int mt = gw >> 4, nt = gw & 15;
        v8f acc = {0.f,0.f,0.f,0.f,0.f,0.f,0.f,0.f};
        acc = mm_tile(zbf, Hd, uhi, Hd, mt * 16, nt * 16, Hd, lane, acc);
        int gn = nt * 16 + (lane & 15);
        int mb = mt * 16 + ((lane >> 4) << 3);
        float bias = b_h[i * Hd + gn];
#pragma unroll
        for (int r = 0; r < 8; ++r) {
          int idx = (mb + r) * Hd + gn;
          float ht = tanhf(hxb[idx] + acc[r] + bias);
          float fv = fbuf[idx], sv = si[idx];
          float ns = fv * sv + (1.f - fv) * ht;
          si[idx] = ns; sbi[idx] = (__bf16)ns; xbuf[idx] += ns;
        }
      }
      gbar(sync);

      // S5: xn2 = LN(x)
      if (gw < Bd) ln_row(xbuf, xnb, gw, lane);
      gbar(sync);

      // S6: u = gelu(xn2 @ up^T)   [64,1024] -> 256 tiles / 128 waves
      for (int tt = gw; tt < 256; tt += NWAVES) {
        int mt = tt >> 6, nt = tt & 63;
        v8f acc = {0.f,0.f,0.f,0.f,0.f,0.f,0.f,0.f};
        acc = mm_tile(xnb, Hd, upi, Hd, mt * 16, nt * 16, Hd, lane, acc);
        int gn = nt * 16 + (lane & 15);
        int mb = mt * 16 + ((lane >> 4) << 3);
#pragma unroll
        for (int r = 0; r < 8; ++r) {
          float v = acc[r];
          ubuf[(mb + r) * H4 + gn] =
              (__bf16)(0.5f * v * (1.f + erff(v * 0.70710678118654752f)));
        }
      }
      gbar(sync);

      // S7: x += u @ down^T   (K = 1024)
      if (gw < 64) {
        int mt = gw >> 4, nt = gw & 15;
        v8f acc = {0.f,0.f,0.f,0.f,0.f,0.f,0.f,0.f};
        acc = mm_tile(ubuf, H4, dni, H4, mt * 16, nt * 16, H4, lane, acc);
        int gn = nt * 16 + (lane & 15);
        int mb = mt * 16 + ((lane >> 4) << 3);
#pragma unroll
        for (int r = 0; r < 8; ++r) xbuf[(mb + r) * Hd + gn] += acc[r];
      }
      gbar(sync);
    }

    // S8: xn3 = LN(x)
    if (gw < Bd) ln_row(xbuf, xnc, gw, lane);
    gbar(sync);

    // S9: logits[b, t, :] = xn3 @ head^T  (written straight to d_out)
    if (gw < 64) {
      int mt = gw >> 4, nt = gw & 15;
      v8f acc = {0.f,0.f,0.f,0.f,0.f,0.f,0.f,0.f};
      acc = mm_tile(xnc, Hd, hd, Hd, mt * 16, nt * 16, Hd, lane, acc);
      int gn = nt * 16 + (lane & 15);
      int mb = mt * 16 + ((lane >> 4) << 3);
#pragma unroll
      for (int r = 0; r < 8; ++r)
        out[(size_t)(mb + r) * Td * Vd + (size_t)t * Vd + gn] = acc[r];
    }
    // next S0 touches only xbuf (last read before S8's barrier): no gbar needed
  }

  gbar(sync);
  // final_states appended after the logits block
  const size_t logits_n = (size_t)Bd * Td * Vd;
  for (int e = gtid; e < Ld * Bd * Hd; e += stride)
    out[logits_n + e] = sbuf[e];
}

extern "C" void kernel_launch(void* const* d_in, const int* in_sizes, int n_in,
                              void* d_out, int out_size, void* d_ws, size_t ws_size,
                              hipStream_t stream) {
  const int*   tokens      = (const int*)d_in[0];
  const float* emb         = (const float*)d_in[1];
  const float* W_f         = (const float*)d_in[2];
  const float* U_f         = (const float*)d_in[3];
  const float* b_f         = (const float*)d_in[4];
  const float* W_h         = (const float*)d_in[5];
  const float* U_h         = (const float*)d_in[6];
  const float* b_h         = (const float*)d_in[7];
  const float* mlp_up      = (const float*)d_in[8];
  const float* mlp_down    = (const float*)d_in[9];
  const float* head        = (const float*)d_in[10];
  const float* init_states = (const float*)d_in[11];

  // barrier control words must start from a known state each launch
  sync_init_kernel<<<1, 64, 0, stream>>>((unsigned*)d_ws);
  mingru_persistent<<<NWG, NTHREADS, 0, stream>>>(
      tokens, emb, W_f, U_f, b_f, W_h, U_h, b_h, mlp_up, mlp_down, head,
      init_states, (float*)d_out, (char*)d_ws);
}
